// SequenceHMM_33938831573605
// MI455X (gfx1250) — matrix operand order repaired and checked by reading, not compiled, not run on hardware
//
#include <hip/hip_runtime.h>

// ---------------------------------------------------------------------------
// HMM forward scan for MI455X (gfx1250, wave32, WMMA).
//   B=64, T=1024, S=512, O=32000
// Per step: alpha_new = m + log( exp(alpha - m) @ P ) + log_obs_gather
// GEMM via v_wmma_f32_16x16x32_bf16.
// B operand split: K=0..223 register-resident (7 frags, 56 VGPRs),
//                  K=224..511 in LDS (303KB of the CDNA5 320KB LDS).
// w tile staged in LDS laid out as WMMA A fragments.
// ---------------------------------------------------------------------------

#define Sdim 512
#define Tdim 1024
#define Odim 32000
#define NRES 7          // register-resident K-steps (7*32 = 224 K values)
#define PH_ELEMS 288    // LDS Pt rows hold K=224..511 (9*32 = 288 elements)
#define PH_STRIDE 296   // padded row stride (elements); 592B = 16*37, 16B-aligned
#define WL_STRIDE 520   // LDS w row stride (elements), padded

typedef __attribute__((ext_vector_type(16))) __bf16 v16bf;
typedef __attribute__((ext_vector_type(8)))  float  v8f;

union ABFrag { uint4 u[2]; v16bf v; };

__device__ __forceinline__ unsigned short f2bf(float f) {
    unsigned u = __float_as_uint(f);
    u += 0x7fffu + ((u >> 16) & 1u);          // round-to-nearest-even
    return (unsigned short)(u >> 16);
}

// order-preserving float<->uint key for atomicMax on LDS
__device__ __forceinline__ unsigned okey(float f) {
    unsigned u = __float_as_uint(f);
    return (u & 0x80000000u) ? ~u : (u | 0x80000000u);
}
__device__ __forceinline__ float unkey(unsigned k) {
    unsigned u = (k & 0x80000000u) ? (k & 0x7fffffffu) : ~k;
    return __uint_as_float(u);
}

// A fragment (16x32 bf16): lanes 0-15 -> M=0..15, K = kb+0..15
//                          lanes 16-31 -> M=0..15, K = kb+16..31
__device__ __forceinline__ v16bf lds_a(const unsigned short* wl, int n, int kb, int g) {
    const uint4* pa = (const uint4*)(wl + n * WL_STRIDE + kb + 16 * g);
    ABFrag t;
    t.u[0] = pa[0];
    t.u[1] = pa[1];   // +8 elements = +16 bytes
    return t.v;
}

// ---------------------------------------------------------------------------
// Precompute: per-row logsumexp of x[R][C] -> out[R]
// ---------------------------------------------------------------------------
__global__ void lse_rows_kernel(const float* __restrict__ x, int C,
                                float* __restrict__ out) {
    __shared__ float red[256];
    const int row = blockIdx.x;
    const float* p = x + (size_t)row * C;
    float mx = -INFINITY;
    for (int c = threadIdx.x; c < C; c += 256) mx = fmaxf(mx, p[c]);
    red[threadIdx.x] = mx;
    __syncthreads();
    for (int s = 128; s > 0; s >>= 1) {
        if (threadIdx.x < s) red[threadIdx.x] = fmaxf(red[threadIdx.x], red[threadIdx.x + s]);
        __syncthreads();
    }
    mx = red[0];
    __syncthreads();
    float sm = 0.f;
    for (int c = threadIdx.x; c < C; c += 256) sm += __expf(p[c] - mx);
    red[threadIdx.x] = sm;
    __syncthreads();
    for (int s = 128; s > 0; s >>= 1) {
        if (threadIdx.x < s) red[threadIdx.x] += red[threadIdx.x + s];
        __syncthreads();
    }
    if (threadIdx.x == 0) out[row] = mx + __logf(red[0]);
}

// ---------------------------------------------------------------------------
// Precompute: Pt[j][i] = softmax_row_i(trans)[j] in bf16 (transposed for B op)
// ---------------------------------------------------------------------------
__global__ void build_pt_kernel(const float* __restrict__ trans,
                                const float* __restrict__ lse_trans,
                                unsigned short* __restrict__ Pt) {
    int idx = blockIdx.x * 256 + threadIdx.x;   // 0 .. 512*512-1
    int j = idx >> 9;
    int i = idx & 511;
    float v = __expf(trans[(size_t)i * Sdim + j] - lse_trans[i]);
    Pt[(size_t)j * Sdim + i] = f2bf(v);         // coalesced writes along i
}

// ---------------------------------------------------------------------------
// Main persistent scan kernel: 4 blocks x 1024 threads (32 waves).
// Block owns batch rows [b0, b0+16). Wave owns 16 output columns.
// ---------------------------------------------------------------------------
__global__ __launch_bounds__(1024) void hmm_scan_kernel(
    const int* __restrict__ obs,              // [B, T]
    const float* __restrict__ obsT,           // [S, O] raw logits
    const float* __restrict__ prior,          // [S]
    const unsigned short* __restrict__ Pt,    // [S, S] bf16, transposed probs
    const float* __restrict__ lse_obs,        // [S]
    const float* __restrict__ lse_prior,      // [1]
    float* __restrict__ out)                  // [B]
{
    extern __shared__ unsigned char smem[];
    unsigned short* ph = (unsigned short*)smem;                          // [512][296] bf16: Pt K=224..511
    unsigned short* wl = (unsigned short*)(smem + Sdim * PH_STRIDE * 2); // [16][520] bf16: A operand
    unsigned* mbuf = (unsigned*)(smem + Sdim * PH_STRIDE * 2 + 16 * WL_STRIDE * 2); // [2][16]
    int*   obs_l = (int*)(mbuf + 32);                                    // [16]
    float* psum  = (float*)(obs_l + 16);                                 // [32][16]

    const int tid  = threadIdx.x;
    const int wave = tid >> 5;
    const int lane = tid & 31;
    const int n    = lane & 15;
    const int g    = lane >> 4;
    const int jcol = wave * 16 + n;          // this lane's output column
    const int b0   = blockIdx.x * 16;

    if (tid < 16)      obs_l[tid] = obs[(size_t)(b0 + tid) * Tdim];
    else if (tid < 48) mbuf[tid - 16] = 0u;

    // cooperative load: Pt[:, K=224..511] -> LDS (303 KB), 36 chunks of 8/row
    for (int it = tid; it < 512 * 36; it += 1024) {
        int j = it / 36, ch = it % 36;
        const uint4* src = (const uint4*)(Pt + (size_t)j * Sdim + (Sdim - PH_ELEMS) + ch * 8);
        *(uint4*)(ph + j * PH_STRIDE + ch * 8) = *src;
    }

    // register-resident B fragments for K=0..223 (7 frags x 8 VGPRs)
    // B (32x16 bf16): lanes 0-15 col n K=kb..kb+15, lanes 16-31 K=kb+16..kb+31
    v16bf Breg[NRES];
    #pragma unroll
    for (int kk = 0; kk < NRES; ++kk) {
        const uint4* pb = (const uint4*)(Pt + (size_t)jcol * Sdim + kk * 32 + 16 * g);
        ABFrag t; t.u[0] = pb[0]; t.u[1] = pb[1];
        Breg[kk] = t.v;
    }

    const float lse_j = lse_obs[jcol];
    const float lp    = prior[jcol] - lse_prior[0];

    __syncthreads();

    // alpha0[b, j] = log_prior[j] + log_obs[j, o_0(b)]
    // D-matrix mapping: lane (n,g), reg r -> row (r + 8g), col jcol
    float alpha[8];
    #pragma unroll
    for (int r = 0; r < 8; ++r) {
        int o = obs_l[r + 8 * g];
        alpha[r] = lp + obsT[(size_t)jcol * Odim + o] - lse_j;
    }

    #pragma unroll 1
    for (int t = 1; t <= Tdim; ++t) {
        const int p = t & 1;
        // row-max across all 512 columns: shuffle tree within 16-lane group,
        // then one LDS atomicMax per (wave, row)
        #pragma unroll
        for (int r = 0; r < 8; ++r) {
            float v = alpha[r];
            v = fmaxf(v, __shfl_xor(v, 1));
            v = fmaxf(v, __shfl_xor(v, 2));
            v = fmaxf(v, __shfl_xor(v, 4));
            v = fmaxf(v, __shfl_xor(v, 8));
            if (n == 0) atomicMax(&mbuf[p * 16 + r + 8 * g], okey(v));
        }
        if (tid < 16)      obs_l[tid] = (t < Tdim) ? obs[(size_t)(b0 + tid) * Tdim + t] : 0;
        else if (tid < 32) mbuf[(p ^ 1) * 16 + (tid - 16)] = 0u;  // reset other buffer
        __syncthreads();

        float m[8];
        #pragma unroll
        for (int r = 0; r < 8; ++r) {
            m[r] = unkey(mbuf[p * 16 + r + 8 * g]);
            wl[(r + 8 * g) * WL_STRIDE + jcol] = f2bf(__expf(alpha[r] - m[r]));
        }
        // prefetch emission gathers; L2 latency hides under the WMMA loop
        float gv[8];
        #pragma unroll
        for (int r = 0; r < 8; ++r) {
            gv[r] = obsT[(size_t)jcol * Odim + obs_l[r + 8 * g]];
        }
        __syncthreads();

        // GEMM: pred_linear[16x16] += w[16x512] @ P[512x16]
        v8f acc = {};
        #pragma unroll
        for (int kk = 0; kk < NRES; ++kk) {
            v16bf a = lds_a(wl, n, kk * 32, g);
            acc = __builtin_amdgcn_wmma_f32_16x16x32_bf16(
                false, a, false, Breg[kk], (short)0, acc, false, false);
        }
        #pragma unroll
        for (int kk = NRES; kk < 16; ++kk) {
            v16bf a = lds_a(wl, n, kk * 32, g);
            const uint4* pb = (const uint4*)(ph + jcol * PH_STRIDE + (kk - NRES) * 32 + 16 * g);
            ABFrag tb; tb.u[0] = pb[0]; tb.u[1] = pb[1];
            acc = __builtin_amdgcn_wmma_f32_16x16x32_bf16(
                false, a, false, tb.v, (short)0, acc, false, false);
        }

        // alpha_new = m + log(acc) + (obsT[j, o_t(b)] - lse_obs[j])
        #pragma unroll
        for (int r = 0; r < 8; ++r) {
            alpha[r] = m[r] + __logf(acc[r]) + gv[r] - lse_j;
        }
    }

    // final logsumexp over states, deterministic reduction
    const int pf = (Tdim + 1) & 1;   // == 1, reset during t == Tdim
    #pragma unroll
    for (int r = 0; r < 8; ++r) {
        float v = alpha[r];
        v = fmaxf(v, __shfl_xor(v, 1));
        v = fmaxf(v, __shfl_xor(v, 2));
        v = fmaxf(v, __shfl_xor(v, 4));
        v = fmaxf(v, __shfl_xor(v, 8));
        if (n == 0) atomicMax(&mbuf[pf * 16 + r + 8 * g], okey(v));
    }
    __syncthreads();
    #pragma unroll
    for (int r = 0; r < 8; ++r) {
        float mm = unkey(mbuf[pf * 16 + r + 8 * g]);
        float e = __expf(alpha[r] - mm);
        e += __shfl_xor(e, 1);
        e += __shfl_xor(e, 2);
        e += __shfl_xor(e, 4);
        e += __shfl_xor(e, 8);
        if (n == 0) psum[wave * 16 + r + 8 * g] = e;   // per-wave partial
    }
    __syncthreads();
    if (tid < 16) {
        float s = 0.f;
        for (int wv = 0; wv < 32; ++wv) s += psum[wv * 16 + tid];
        out[b0 + tid] = unkey(mbuf[pf * 16 + tid]) + __logf(s);
    }
}

// ---------------------------------------------------------------------------
extern "C" void kernel_launch(void* const* d_in, const int* in_sizes, int n_in,
                              void* d_out, int out_size, void* d_ws, size_t ws_size,
                              hipStream_t stream) {
    (void)in_sizes; (void)n_in; (void)out_size; (void)ws_size;
    const int*   obs   = (const int*)d_in[0];     // [64, 1024]
    const float* trans = (const float*)d_in[1];   // [512, 512]
    const float* obsT  = (const float*)d_in[2];   // [512, 32000]
    const float* prior = (const float*)d_in[3];   // [512]
    float* out = (float*)d_out;                   // [64]

    char* ws = (char*)d_ws;
    unsigned short* Pt  = (unsigned short*)ws;              // 512*512*2 = 524288 B
    float* lse_obs_p    = (float*)(ws + 524288);            // 2048 B
    float* lse_trans_p  = (float*)(ws + 526336);            // 2048 B
    float* lse_prior_p  = (float*)(ws + 528384);            // 4 B

    lse_rows_kernel<<<512, 256, 0, stream>>>(obsT, Odim, lse_obs_p);
    lse_rows_kernel<<<512, 256, 0, stream>>>(trans, Sdim, lse_trans_p);
    lse_rows_kernel<<<1, 256, 0, stream>>>(prior, Sdim, lse_prior_p);
    build_pt_kernel<<<(Sdim * Sdim) / 256, 256, 0, stream>>>(trans, lse_trans_p, Pt);

    const size_t smem_bytes =
        (size_t)Sdim * PH_STRIDE * 2 +   // ph    303104
        (size_t)16 * WL_STRIDE * 2 +     // wl     16640
        32 * sizeof(unsigned) +          // mbuf     128
        16 * sizeof(int) +               // obs_l     64
        32 * 16 * sizeof(float);         // psum    2048  -> 321984 <= 327680
    hmm_scan_kernel<<<4, 1024, smem_bytes, stream>>>(
        obs, obsT, prior, Pt, lse_obs_p, lse_prior_p, out);
}
